// MultiHeadAttention_30013231464690
// MI455X (gfx1250) — compile-verified, matrix operand order checked by reading, not run on hardware
//
#include <hip/hip_runtime.h>
#include <cstdint>
#include <cstddef>

// ---------------------------------------------------------------------------
// MultiHeadAttention for MI455X (gfx1250), bf16 WMMA with fp32 accumulation.
//   B=8, S=2048, E=1024, H=8, D=128
// Pipeline:
//   1) cvt_f32_bf16:  x, Wq, Wk, Wv (stacked [1024,1024]), Wo -> bf16 in ws
//   2) gemm_bf16_wmma x3: Q/K/V = x @ W^T + b   -> bf16 [B,H,S,D]
//   3) flash_attn: online-softmax attention     -> bf16 concat [BS, E]
//   4) gemm_bf16_wmma: concat @ Wo^T + bo       -> f32  d_out
// Workspace requirement: ~168 MB.
// ---------------------------------------------------------------------------

#define BATCH 8
#define SEQ   2048
#define EMBED 1024
#define HEADS 8
#define HDIM  128
#define BS    (BATCH * SEQ)   /* 16384 */
#define GK    1024            /* GEMM reduction dim */
#define GN    1024            /* GEMM N dim */

typedef __bf16        v16bf __attribute__((ext_vector_type(16)));
typedef float         v8f   __attribute__((ext_vector_type(8)));
typedef unsigned int  v4u   __attribute__((ext_vector_type(4)));

union Frag16 { v4u u[2]; v16bf v; };           // 32 bytes: one 16-bit A/B fragment

__device__ __forceinline__ unsigned short f2bf(float x) {
  union { float f; unsigned u; } c; c.f = x;
  unsigned r = c.u + 0x7FFFu + ((c.u >> 16) & 1u);   // round-to-nearest-even
  return (unsigned short)(r >> 16);
}

// ---------------------------------------------------------------------------
// 1) f32 -> bf16 conversion (grid-stride)
// ---------------------------------------------------------------------------
__global__ void cvt_f32_bf16(const float* __restrict__ in,
                             unsigned short* __restrict__ out, int n) {
  int i = blockIdx.x * blockDim.x + threadIdx.x;
  int stride = gridDim.x * blockDim.x;
  for (; i < n; i += stride) out[i] = f2bf(in[i]);
}

// ---------------------------------------------------------------------------
// 2/4) Tiled bf16 GEMM:  C[M=16384, N=1024] = A[M,K=1024] * Bw[N,K]^T + bias
// Block tile 128x128, 8 waves; each wave: 4(M) x 2(N) accumulator frags.
// mode 0: store bf16 scattered to [B,H,S,D] (QKV).  mode 1: store f32 row-major.
// ---------------------------------------------------------------------------
__global__ __launch_bounds__(256)
void gemm_bf16_wmma(const unsigned short* __restrict__ A,    // [BS][GK] bf16
                    const unsigned short* __restrict__ Bw,   // [GN][GK] bf16
                    const float* __restrict__ bias,          // [GN]
                    unsigned short* __restrict__ outQ,       // mode 0
                    float* __restrict__ outF,                // mode 1
                    int mode) {
  __shared__ unsigned short As[128 * 32];   // 8 KB
  __shared__ unsigned short Bs[128 * 32];   // 8 KB

  const int tid  = threadIdx.x;
  const int lane = tid & 31;
  const int wave = tid >> 5;
  const int wm   = wave & 1;     // 2 wave rows   -> 64 M-rows each
  const int wn   = wave >> 1;    // 4 wave cols   -> 32 N-cols each
  const int hi   = lane >> 4;    // half-wave: selects K offset 0/8 in frags
  const int ln   = lane & 15;
  const int m0   = blockIdx.x * 128;
  const int n0   = blockIdx.y * 128;

  const int lrow = tid >> 1;     // 0..127 : tile row this thread stages
  const int lhalf = (tid & 1) * 16;

  v8f acc[4][2];
  const v8f vzero = {0.f, 0.f, 0.f, 0.f, 0.f, 0.f, 0.f, 0.f};
#pragma unroll
  for (int i = 0; i < 4; ++i)
#pragma unroll
    for (int j = 0; j < 2; ++j) acc[i][j] = vzero;

  for (int k0 = 0; k0 < GK; k0 += 32) {
    __syncthreads();   // previous iteration's frag reads done before overwrite
    {
      const v4u* ga = (const v4u*)(A + (size_t)(m0 + lrow) * GK + k0 + lhalf);
      v4u a0 = ga[0], a1 = ga[1];
      const v4u* gb = (const v4u*)(Bw + (size_t)(n0 + lrow) * GK + k0 + lhalf);
      v4u b0 = gb[0], b1 = gb[1];
      v4u* la = (v4u*)(As + lrow * 32 + lhalf);
      la[0] = a0; la[1] = a1;
      v4u* lb = (v4u*)(Bs + lrow * 32 + lhalf);
      lb[0] = b0; lb[1] = b1;
      if (k0 + 32 < GK) {   // uniform guard; lowers to global_prefetch_b8
        __builtin_prefetch(A + (size_t)(m0 + lrow) * GK + k0 + 32 + lhalf, 0, 1);
        __builtin_prefetch(Bw + (size_t)(n0 + lrow) * GK + k0 + 32 + lhalf, 0, 1);
      }
    }
    __syncthreads();

    // Fragment loads per CDNA5 16-bit A/B layout: lane = row, two b128 reads
    // covering K = kb..kb+7 and K = kb+16..kb+23 (kb = 8*hi).
    Frag16 af[4], bf[2];
    const int kb = hi * 8;
#pragma unroll
    for (int mt = 0; mt < 4; ++mt) {
      const unsigned short* p = As + (wm * 64 + mt * 16 + ln) * 32 + kb;
      af[mt].u[0] = *(const v4u*)p;
      af[mt].u[1] = *(const v4u*)(p + 16);
    }
#pragma unroll
    for (int nt = 0; nt < 2; ++nt) {
      const unsigned short* p = Bs + (wn * 32 + nt * 16 + ln) * 32 + kb;
      bf[nt].u[0] = *(const v4u*)p;
      bf[nt].u[1] = *(const v4u*)(p + 16);
    }

#pragma unroll
    for (int mt = 0; mt < 4; ++mt)
#pragma unroll
      for (int nt = 0; nt < 2; ++nt)
        acc[mt][nt] = __builtin_amdgcn_wmma_f32_16x16x32_bf16(
            false, af[mt].v, false, bf[nt].v, (short)0, acc[mt][nt], false, false);
  }

  // Epilogue. C-frag layout: row = v + 8*hi, col = ln.
#pragma unroll
  for (int mt = 0; mt < 4; ++mt) {
#pragma unroll
    for (int nt = 0; nt < 2; ++nt) {
#pragma unroll
      for (int v = 0; v < 8; ++v) {
        const int r = m0 + wm * 64 + mt * 16 + v + 8 * hi;
        const int n = n0 + wn * 32 + nt * 16 + ln;
        const float val = acc[mt][nt][v] + bias[n];
        if (mode == 0) {
          const int b = r >> 11, s = r & 2047;        // S = 2048
          const int h = n >> 7, d = n & 127;          // D = 128
          outQ[(((size_t)(b * HEADS + h) * SEQ + s) << 7) + d] = f2bf(val);
        } else {
          outF[(size_t)r * GN + n] = val;
        }
      }
    }
  }
}

// ---------------------------------------------------------------------------
// 3) Flash attention. Grid = B*H*(S/128); 8 waves, each owns 16 query rows.
//    Streams 32-key tiles through LDS (K row-major, V transposed).
// ---------------------------------------------------------------------------
__global__ __launch_bounds__(256)
void flash_attn(const unsigned short* __restrict__ Q,
                const unsigned short* __restrict__ K,
                const unsigned short* __restrict__ V,
                unsigned short* __restrict__ Cb) {
  __shared__ unsigned short Kt[32 * 128];      // [key][d]  8 KB
  __shared__ unsigned short Vt[128 * 32];      // [d][key]  8 KB (transposed)
  __shared__ unsigned short Pw[8][16 * 32];    // per-wave P scratch, 8 KB

  const int tid  = threadIdx.x;
  const int lane = tid & 31;
  const int wave = tid >> 5;
  const int hi   = lane >> 4;
  const int ln   = lane & 15;
  const int qblk = blockIdx.x & 15;            // S/128 = 16 query blocks
  const int bh   = blockIdx.x >> 4;            // b*H + h
  const size_t base = (size_t)bh * SEQ * HDIM;

  const unsigned short* Qp = Q + base + (size_t)(qblk * 128 + wave * 16) * HDIM;
  const unsigned short* Kp = K + base;
  const unsigned short* Vp = V + base;

  // Q fragments: 16x128 as 4 k32 A-frags, resident for the whole kernel.
  Frag16 qf[4];
#pragma unroll
  for (int kk = 0; kk < 4; ++kk) {
    const unsigned short* p = Qp + ln * HDIM + kk * 32 + hi * 8;
    qf[kk].u[0] = *(const v4u*)p;
    qf[kk].u[1] = *(const v4u*)(p + 16);
  }

  v8f oacc[8];
  const v8f vzero = {0.f, 0.f, 0.f, 0.f, 0.f, 0.f, 0.f, 0.f};
  float mrow[8], lrow[8];
#pragma unroll
  for (int i = 0; i < 8; ++i) { oacc[i] = vzero; mrow[i] = -3.0e38f; lrow[i] = 0.f; }

  const float scale = 0.08838834764831845f;    // 1/sqrt(128)
  const int kvrow  = tid >> 3;                 // 0..31 key row staged
  const int dchunk = (tid & 7) * 16;           // 0..112 d chunk staged

  for (int kt = 0; kt < SEQ / 32; ++kt) {
    __syncthreads();   // prior PV reads of Kt/Vt complete
    {
      const v4u* gk = (const v4u*)(Kp + (size_t)(kt * 32 + kvrow) * HDIM + dchunk);
      v4u k0 = gk[0], k1 = gk[1];
      v4u* lk = (v4u*)(Kt + kvrow * 128 + dchunk);
      lk[0] = k0; lk[1] = k1;

      union { v4u u[2]; unsigned short s[16]; } vv;
      const v4u* gv = (const v4u*)(Vp + (size_t)(kt * 32 + kvrow) * HDIM + dchunk);
      vv.u[0] = gv[0]; vv.u[1] = gv[1];
#pragma unroll
      for (int j = 0; j < 16; ++j)
        Vt[(dchunk + j) * 32 + kvrow] = vv.s[j];   // transpose into LDS
    }
    __syncthreads();

    // Scores: S16x32 = Q(16x128) * K^T, as 2 n-tiles x 4 chained k32 WMMAs.
    v8f sf[2];
    sf[0] = vzero; sf[1] = vzero;
#pragma unroll
    for (int nt = 0; nt < 2; ++nt) {
#pragma unroll
      for (int kk = 0; kk < 4; ++kk) {
        Frag16 bfr;
        const unsigned short* p = Kt + (nt * 16 + ln) * 128 + kk * 32 + hi * 8;
        bfr.u[0] = *(const v4u*)p;
        bfr.u[1] = *(const v4u*)(p + 16);
        sf[nt] = __builtin_amdgcn_wmma_f32_16x16x32_bf16(
            false, qf[kk].v, false, bfr.v, (short)0, sf[nt], false, false);
      }
    }

    // Online softmax; each C-frag row lives in one 16-lane half (xor-reduce).
#pragma unroll
    for (int v = 0; v < 8; ++v) {
      float s0 = sf[0][v] * scale;
      float s1 = sf[1][v] * scale;
      float tmax = fmaxf(s0, s1);
#pragma unroll
      for (int off = 1; off < 16; off <<= 1)
        tmax = fmaxf(tmax, __shfl_xor(tmax, off, 32));
      const float mnew = fmaxf(mrow[v], tmax);
      const float p0 = __expf(s0 - mnew);
      const float p1 = __expf(s1 - mnew);
      const float corr = __expf(mrow[v] - mnew);
      float rs = p0 + p1;
#pragma unroll
      for (int off = 1; off < 16; off <<= 1)
        rs += __shfl_xor(rs, off, 32);
      lrow[v] = lrow[v] * corr + rs;
      mrow[v] = mnew;
#pragma unroll
      for (int dt = 0; dt < 8; ++dt) oacc[dt][v] *= corr;
      const int prow = v + 8 * hi;               // C layout -> LDS re-layout
      Pw[wave][prow * 32 + ln]      = f2bf(p0);
      Pw[wave][prow * 32 + 16 + ln] = f2bf(p1);
    }
    __syncthreads();

    // O += P(16x32) * V(32x128): A-frag from Pw, B-frag from transposed Vt.
    Frag16 pa;
    {
      const unsigned short* pp = &Pw[wave][ln * 32 + hi * 8];
      pa.u[0] = *(const v4u*)pp;
      pa.u[1] = *(const v4u*)(pp + 16);
    }
#pragma unroll
    for (int dt = 0; dt < 8; ++dt) {
      Frag16 vb;
      const unsigned short* vp = Vt + (dt * 16 + ln) * 32 + hi * 8;
      vb.u[0] = *(const v4u*)vp;
      vb.u[1] = *(const v4u*)(vp + 16);
      oacc[dt] = __builtin_amdgcn_wmma_f32_16x16x32_bf16(
          false, pa.v, false, vb.v, (short)0, oacc[dt], false, false);
    }
  }

  // Normalize and store concat layout [b, s, h*128 + d] as bf16.
  const int h = bh & 7, b = bh >> 3;
#pragma unroll
  for (int v = 0; v < 8; ++v) {
    const float inv = 1.0f / lrow[v];
    const int r = (b << 11) + qblk * 128 + wave * 16 + v + 8 * hi;
    const size_t rowoff = (size_t)r * EMBED + h * HDIM;
#pragma unroll
    for (int dt = 0; dt < 8; ++dt)
      Cb[rowoff + dt * 16 + ln] = f2bf(oacc[dt][v] * inv);
  }
}

// ---------------------------------------------------------------------------
// Launcher
// ---------------------------------------------------------------------------
extern "C" void kernel_launch(void* const* d_in, const int* in_sizes, int n_in,
                              void* d_out, int out_size, void* d_ws, size_t ws_size,
                              hipStream_t stream) {
  const float* x  = (const float*)d_in[0];
  const float* Wq = (const float*)d_in[1];
  const float* bq = (const float*)d_in[2];
  const float* Wk = (const float*)d_in[3];
  const float* bk = (const float*)d_in[4];
  const float* Wv = (const float*)d_in[5];
  const float* bv = (const float*)d_in[6];
  const float* Wo = (const float*)d_in[7];
  const float* bo = (const float*)d_in[8];
  float* out = (float*)d_out;

  char* ws = (char*)d_ws;
  unsigned short* xb  = (unsigned short*)ws;  ws += (size_t)BS * EMBED * 2;    // 32 MB
  unsigned short* wqb = (unsigned short*)ws;  ws += (size_t)GN * GK * 2;       //  2 MB
  unsigned short* wkb = (unsigned short*)ws;  ws += (size_t)GN * GK * 2;
  unsigned short* wvb = (unsigned short*)ws;  ws += (size_t)GN * GK * 2;
  unsigned short* wob = (unsigned short*)ws;  ws += (size_t)GN * GK * 2;
  unsigned short* Qb  = (unsigned short*)ws;  ws += (size_t)BS * EMBED * 2;    // 32 MB
  unsigned short* Kb  = (unsigned short*)ws;  ws += (size_t)BS * EMBED * 2;
  unsigned short* Vb  = (unsigned short*)ws;  ws += (size_t)BS * EMBED * 2;
  unsigned short* Cbf = (unsigned short*)ws;  ws += (size_t)BS * EMBED * 2;

  // 1) convert fp32 inputs to bf16
  cvt_f32_bf16<<<8192, 256, 0, stream>>>(x,  xb,  BS * EMBED);
  cvt_f32_bf16<<<2048, 256, 0, stream>>>(Wq, wqb, GN * GK);
  cvt_f32_bf16<<<2048, 256, 0, stream>>>(Wk, wkb, GN * GK);
  cvt_f32_bf16<<<2048, 256, 0, stream>>>(Wv, wvb, GN * GK);
  cvt_f32_bf16<<<2048, 256, 0, stream>>>(Wo, wob, GN * GK);

  // 2) QKV projections -> bf16 [B,H,S,D]
  dim3 ggrid(BS / 128, GN / 128);   // 128 x 8
  gemm_bf16_wmma<<<ggrid, 256, 0, stream>>>(xb, wqb, bq, Qb, nullptr, 0);
  gemm_bf16_wmma<<<ggrid, 256, 0, stream>>>(xb, wkb, bk, Kb, nullptr, 0);
  gemm_bf16_wmma<<<ggrid, 256, 0, stream>>>(xb, wvb, bv, Vb, nullptr, 0);

  // 3) attention -> bf16 concat [BS, E]
  flash_attn<<<BATCH * HEADS * (SEQ / 128), 256, 0, stream>>>(Qb, Kb, Vb, Cbf);

  // 4) output projection -> fp32 d_out
  gemm_bf16_wmma<<<ggrid, 256, 0, stream>>>(Cbf, wob, bo, nullptr, out, 1);
}
